// SwinTransformer_54563264529127
// MI455X (gfx1250) — matrix operand order repaired and checked by reading, not compiled, not run on hardware
//
#include <hip/hip_runtime.h>
#include <cstdint>

typedef _Float16 half_t;
typedef __attribute__((ext_vector_type(16))) _Float16 v16h;
typedef __attribute__((ext_vector_type(8)))  float    v8f;

#define IMG    56
#define CH     256
#define WSZ    7
#define NTOK   49      // tokens per window (7*7)
#define NHEAD  8
#define HD     32
#define SHF    3
#define QKV_NT 48      // 768/16 column tiles
#define PRJ_NT 16      // 256/16 column tiles
#define PK_QKV_DW (QKV_NT*8*32*8)   // ntile * kstep * lane * dword
#define PK_PRJ_DW (PRJ_NT*8*32*8)
#define ADD_N    (4*NHEAD*4*4*32*8) // wtype * head * mt * nt * lane * r = 131072

// ---------------------------------------------------------------- helpers
__device__ __forceinline__ v8f wmma_f16(v16h a, v16h b, v8f c) {
  return __builtin_amdgcn_wmma_f32_16x16x32_f16(
      /*neg_a=*/false, a, /*neg_b=*/false, b,
      /*c_mod=*/(short)0, c, /*reuse_a=*/false, /*reuse_b=*/false);
}

__device__ __forceinline__ v8f splat8(float x) {
  v8f v;
#pragma unroll
  for (int i = 0; i < 8; ++i) v[i] = x;
  return v;
}

// A-fragment (16x32 f16) from row-major f16 LDS: lane holds row = base_row,
// K chunk {koff..koff+15} (koff already includes the
// lane-half offset of 16 for lanes 16..31).
__device__ __forceinline__ v16h load_a_frag(const half_t* base, int stride_h,
                                            int row, int koff) {
  union { uint4 u[2]; v16h h; } f;
  const half_t* p = base + row * stride_h + koff;
  f.u[0] = *(const uint4*)(p);
  f.u[1] = *(const uint4*)(p + 8);
  return f.h;
}

// B-fragment from 16 contiguous f16 (K kbase..kbase+15 at fixed column).
__device__ __forceinline__ v16h load_b_contig(const half_t* p) {
  union { uint4 u[2]; v16h h; } f;
  f.u[0] = *(const uint4*)(p);
  f.u[1] = *(const uint4*)(p + 8);
  return f.h;
}

// B-fragment from pre-packed global weights (8 consecutive dwords per lane).
__device__ __forceinline__ v16h load_b_packed(const uint32_t* p) {
  union { uint4 u[2]; v16h h; } f;
  f.u[0] = *(const uint4*)(p);
  f.u[1] = *(const uint4*)(p + 4);
  return f.h;
}

// 8 consecutive f32 (D-fragment-ordered addend table).
__device__ __forceinline__ v8f load_v8f(const float* p) {
  union { float4 f[2]; v8f v; } u;
  u.f[0] = *(const float4*)(p);
  u.f[1] = *(const float4*)(p + 4);
  return u.v;
}

__device__ __forceinline__ int regio(int p) {        // shift-mask region id
  return p < (IMG - WSZ) ? 0 : (p < (IMG - SHF) ? 1 : 2);
}

// ---------------------------------------------------------------- prep
// 1) Pack qkv_weight [768,256] and proj_weight [256,256] (f32, row-major,
//    row = output feature) into f16 B-operand VGPR order:
//      dword index = ((ntile*8 + kstep)*32 + lane)*8 + v
//      holds B[k][n], B[k+1][n] with n = ntile*16 + lane%16,
//      k = kstep*32 + (lane<16 ? 0 : 16) + 2v,  B[k][n] = W[n][k]
// 2) Build combined attention addend table (relpos bias + shift mask +
//    -1e30 column padding) in WMMA D-fragment order:
//      [wtype][head][mt][nt][lane][r], wtype = (wh==7)*2 + (ww==7)
__global__ __launch_bounds__(256) void prep_tables(
    const float* __restrict__ qkv_w, const float* __restrict__ proj_w,
    const float* __restrict__ rel_tab,
    uint32_t* __restrict__ pk_qkv, uint32_t* __restrict__ pk_proj,
    float* __restrict__ addend) {
  int idx = blockIdx.x * 256 + threadIdx.x;
  if (idx < PK_QKV_DW + PK_PRJ_DW) {
    const float* w;
    uint32_t* dst;
    int j;
    if (idx < PK_QKV_DW) { w = qkv_w; dst = pk_qkv; j = idx; }
    else                 { w = proj_w; dst = pk_proj; j = idx - PK_QKV_DW; }
    int v    = j & 7;
    int lane = (j >> 3) & 31;
    int ks   = (j >> 8) & 7;
    int nt   = j >> 11;
    int n  = nt * 16 + (lane & 15);
    int k0 = ks * 32 + ((lane < 16) ? 0 : 16) + 2 * v;
    union { half_t h[2]; uint32_t u; } p;
    p.h[0] = (half_t)w[n * CH + k0];
    p.h[1] = (half_t)w[n * CH + k0 + 1];
    dst[j] = p.u;
  } else if (idx < PK_QKV_DW + PK_PRJ_DW + ADD_N) {
    int j    = idx - (PK_QKV_DW + PK_PRJ_DW);
    int r    = j & 7;
    int lane = (j >> 3) & 31;
    int nt   = (j >> 8) & 3;
    int mt   = (j >> 10) & 3;
    int h    = (j >> 12) & 7;
    int wt   = j >> 15;                      // 0..3
    int m = mt * 16 + (lane >> 4) * 8 + r;
    int n = nt * 16 + (lane & 15);
    float a;
    if (n >= NTOK) {
      a = -1e30f;                            // pad columns out of softmax
    } else {
      int mm = (m < NTOK) ? m : 0;
      int mi = mm / 7, mj = mm - 7 * mi;
      int ni = n / 7,  nj = n - 7 * ni;
      int wh = (wt & 2) ? 7 : 0;             // edge windows sit at grid 7
      int ww = (wt & 1) ? 7 : 0;
      int cm = regio(wh * WSZ + mi) * 3 + regio(ww * WSZ + mj);
      int cn = regio(wh * WSZ + ni) * 3 + regio(ww * WSZ + nj);
      int ridx = (mi - ni + 6) * 13 + (mj - nj + 6);
      a = rel_tab[ridx * NHEAD + h] + ((cm != cn) ? -100.f : 0.f);
    }
    addend[j] = a;
  }
}

// ---------------------------------------------------------------- fused window
__global__ __launch_bounds__(256, 1) void swin_window_attn(
    const float* __restrict__ x,          // [64,56,56,256]
    const float* __restrict__ qkv_bias,   // [768]
    const float* __restrict__ proj_bias,  // [256]
    const uint32_t* __restrict__ pk_qkv,
    const uint32_t* __restrict__ pk_proj,
    const float* __restrict__ addend,     // [4][8][4][4][32][8]
    float* __restrict__ out) {            // [64,56,56,256]
  extern __shared__ __align__(16) char smem_raw[];
  half_t* smA = (half_t*)smem_raw;        // [64][256] window acts, later concat O

  const int tid   = threadIdx.x;
  const int lane  = tid & 31;
  const int wv    = tid >> 5;             // wave id == head id
  const int ln15  = lane & 15;
  const int lhalf = lane >> 4;            // 0/1

  half_t* smQ = (half_t*)(smem_raw + 32 * 1024 + wv * 12 * 1024); // [64][32]
  half_t* smK = smQ + 64 * 32;                                    // [64][32]
  half_t* smP = smQ;                                              // [64][64] overlays Q+K
  half_t* smV = smQ + 2 * 64 * 32;                                // [32][64] = V^T

  const int widx = blockIdx.x;            // b*64 + wh*8 + ww
  const int b  = widx >> 6;
  const int wh = (widx >> 3) & 7;
  const int ww = widx & 7;

  // ---- gather shifted window -> f16 LDS (wide: f32x4 -> f16x4), pad 0 --
  {
    const int sub   = tid & 63;           // 64 threads cover a 256-ch row
    const int rbase = tid >> 6;           // 0..3
#pragma unroll
    for (int it = 0; it < 16; ++it) {
      int r = it * 4 + rbase;             // 0..63
      float4 v4 = make_float4(0.f, 0.f, 0.f, 0.f);
      if (r < NTOK) {
        int i = r / 7, j = r - 7 * (r / 7);
        int sh = wh * WSZ + i + SHF; if (sh >= IMG) sh -= IMG;
        int sw = ww * WSZ + j + SHF; if (sw >= IMG) sw -= IMG;
        v4 = *(const float4*)(x + (((size_t)b * IMG + sh) * IMG + sw) * CH + sub * 4);
      }
      union { half_t h[4]; uint2 u; } p;
      p.h[0] = (half_t)v4.x; p.h[1] = (half_t)v4.y;
      p.h[2] = (half_t)v4.z; p.h[3] = (half_t)v4.w;
      *(uint2*)(smA + r * CH + sub * 4) = p.u;
    }
  }
  __syncthreads();

  const float qscale = 0.17677669529663687f;  // hd^-0.5

  // ---- QKV GEMM: wave computes its head's q,k,v (49x32 each) ----------
#pragma unroll
  for (int s = 0; s < 3; ++s) {               // 0=q 1=k 2=v
    const int ntg = s * 16 + 2 * wv;          // column-tile base within 48
    float b0 = qkv_bias[s * CH + wv * HD + ln15];
    float b1 = qkv_bias[s * CH + wv * HD + 16 + ln15];
    v8f acc[4][2];
#pragma unroll
    for (int mt = 0; mt < 4; ++mt) { acc[mt][0] = splat8(b0); acc[mt][1] = splat8(b1); }
#pragma unroll
    for (int ks = 0; ks < 8; ++ks) {
      v16h bfr[2], afr[4];
#pragma unroll
      for (int nt = 0; nt < 2; ++nt)
        bfr[nt] = load_b_packed(pk_qkv + ((size_t)((ntg + nt) * 8 + ks) * 32 + lane) * 8);
#pragma unroll
      for (int mt = 0; mt < 4; ++mt)
        afr[mt] = load_a_frag(smA, CH, mt * 16 + ln15, ks * 32 + lhalf * 16);
#pragma unroll
      for (int mt = 0; mt < 4; ++mt)
#pragma unroll
        for (int nt = 0; nt < 2; ++nt)
          acc[mt][nt] = wmma_f16(afr[mt], bfr[nt], acc[mt][nt]);
    }
    // spill D-layout accumulators to per-wave LDS
    if (s == 2) {
      // V stored transposed: adjacent rows are contiguous -> packed b32
#pragma unroll
      for (int mt = 0; mt < 4; ++mt)
#pragma unroll
        for (int nt = 0; nt < 2; ++nt)
#pragma unroll
          for (int r = 0; r < 8; r += 2) {
            int m = mt * 16 + lhalf * 8 + r;
            int n = nt * 16 + ln15;
            union { half_t h[2]; uint32_t u; } pp;
            pp.h[0] = (half_t)acc[mt][nt][r];
            pp.h[1] = (half_t)acc[mt][nt][r + 1];
            *(uint32_t*)(smV + n * 64 + m) = pp.u;
          }
    } else {
      half_t* dstq = (s == 0) ? smQ : smK;
      float scl = (s == 0) ? qscale : 1.f;
#pragma unroll
      for (int mt = 0; mt < 4; ++mt)
#pragma unroll
        for (int nt = 0; nt < 2; ++nt)
#pragma unroll
          for (int r = 0; r < 8; ++r) {
            int m = mt * 16 + lhalf * 8 + r;
            int n = nt * 16 + ln15;
            dstq[m * HD + n] = (half_t)(acc[mt][nt][r] * scl);
          }
    }
  }

  // ---- attention: S = q k^T (+precomputed addend), softmax, P -> LDS --
  v16h qfr[4], kfr[4];
#pragma unroll
  for (int t = 0; t < 4; ++t) {
    qfr[t] = load_a_frag(smQ, HD, t * 16 + ln15, lhalf * 16);
    kfr[t] = load_b_contig(smK + (t * 16 + ln15) * HD + lhalf * 16);
  }

  const int wt = ((wh == 7) ? 2 : 0) + ((ww == 7) ? 1 : 0);
  const float* addb = addend + (size_t)(wt * NHEAD + wv) * (4 * 4 * 32 * 8);

#pragma unroll
  for (int mt = 0; mt < 4; ++mt) {
    v8f sacc[4];
#pragma unroll
    for (int nt = 0; nt < 4; ++nt)
      sacc[nt] = wmma_f16(qfr[mt], kfr[nt], splat8(0.f));

    // D-fragment-ordered addend: 2x b128 per 16x16 tile
#pragma unroll
    for (int nt = 0; nt < 4; ++nt) {
      v8f ad = load_v8f(addb + ((mt * 4 + nt) * 32 + lane) * 8);
      sacc[nt] += ad;
    }

    // rowwise softmax: rows live at fixed vgpr index r across the 16-lane half
    float inv[8];
#pragma unroll
    for (int r = 0; r < 8; ++r) {
      float mx = fmaxf(fmaxf(sacc[0][r], sacc[1][r]), fmaxf(sacc[2][r], sacc[3][r]));
      for (int d = 1; d < 16; d <<= 1) mx = fmaxf(mx, __shfl_xor(mx, d, 32));
      float s = 0.f;
#pragma unroll
      for (int nt = 0; nt < 4; ++nt) {
        float e = __expf(sacc[nt][r] - mx);
        sacc[nt][r] = e;
        s += e;
      }
      for (int d = 1; d < 16; d <<= 1) s += __shfl_xor(s, d, 32);
      inv[r] = 1.f / s;
    }
#pragma unroll
    for (int nt = 0; nt < 4; ++nt) {
      int n = nt * 16 + ln15;
#pragma unroll
      for (int r = 0; r < 8; ++r) {
        int m = mt * 16 + lhalf * 8 + r;
        smP[m * 64 + n] = (half_t)(sacc[nt][r] * inv[r]);
      }
    }
  }

  // ---- O = P @ V  (K = 64 padded tokens, N = 32) ----------------------
  v8f oacc[4][2];
#pragma unroll
  for (int mt = 0; mt < 4; ++mt) { oacc[mt][0] = splat8(0.f); oacc[mt][1] = splat8(0.f); }
#pragma unroll
  for (int ks = 0; ks < 2; ++ks) {
    v16h vfr[2], pfr[4];
#pragma unroll
    for (int nt = 0; nt < 2; ++nt)
      vfr[nt] = load_b_contig(smV + (nt * 16 + ln15) * 64 + ks * 32 + lhalf * 16);
#pragma unroll
    for (int mt = 0; mt < 4; ++mt)
      pfr[mt] = load_a_frag(smP, 64, mt * 16 + ln15, ks * 32 + lhalf * 16);
#pragma unroll
    for (int mt = 0; mt < 4; ++mt)
#pragma unroll
      for (int nt = 0; nt < 2; ++nt)
        oacc[mt][nt] = wmma_f16(pfr[mt], vfr[nt], oacc[mt][nt]);
  }

  // ---- concat heads into smA (reuse), then proj GEMM ------------------
  __syncthreads();   // everyone done reading the window activations
#pragma unroll
  for (int mt = 0; mt < 4; ++mt)
#pragma unroll
    for (int nt = 0; nt < 2; ++nt)
#pragma unroll
      for (int r = 0; r < 8; ++r) {
        int m = mt * 16 + lhalf * 8 + r;
        int n = nt * 16 + ln15;
        smA[m * CH + wv * HD + n] = (half_t)oacc[mt][nt][r];
      }
  __syncthreads();

  v8f pacc[4][2];
  {
    float b0 = proj_bias[wv * HD + ln15];
    float b1 = proj_bias[wv * HD + 16 + ln15];
#pragma unroll
    for (int mt = 0; mt < 4; ++mt) { pacc[mt][0] = splat8(b0); pacc[mt][1] = splat8(b1); }
  }
#pragma unroll
  for (int ks = 0; ks < 8; ++ks) {
    v16h bfr[2], afr[4];
#pragma unroll
    for (int nt = 0; nt < 2; ++nt)
      bfr[nt] = load_b_packed(pk_proj + ((size_t)((2 * wv + nt) * 8 + ks) * 32 + lane) * 8);
#pragma unroll
    for (int mt = 0; mt < 4; ++mt)
      afr[mt] = load_a_frag(smA, CH, mt * 16 + ln15, ks * 32 + lhalf * 16);
#pragma unroll
    for (int mt = 0; mt < 4; ++mt)
#pragma unroll
      for (int nt = 0; nt < 2; ++nt)
        pacc[mt][nt] = wmma_f16(afr[mt], bfr[nt], pacc[mt][nt]);
  }

  // ---- scatter back with reverse cyclic shift -------------------------
#pragma unroll
  for (int mt = 0; mt < 4; ++mt)
#pragma unroll
    for (int r = 0; r < 8; ++r) {
      int m = mt * 16 + lhalf * 8 + r;
      if (m < NTOK) {
        int i = m / 7, j = m - 7 * (m / 7);
        int dh = wh * WSZ + i + SHF; if (dh >= IMG) dh -= IMG;
        int dw = ww * WSZ + j + SHF; if (dw >= IMG) dw -= IMG;
        size_t base = (((size_t)b * IMG + dh) * IMG + dw) * CH + wv * HD + ln15;
        out[base]      = pacc[mt][0][r];
        out[base + 16] = pacc[mt][1][r];
      }
    }
}

// ---------------------------------------------------------------- launch
extern "C" void kernel_launch(void* const* d_in, const int* in_sizes, int n_in,
                              void* d_out, int out_size, void* d_ws, size_t ws_size,
                              hipStream_t stream) {
  (void)in_sizes; (void)n_in; (void)out_size; (void)ws_size;
  const float* x      = (const float*)d_in[0];
  const float* qkv_w  = (const float*)d_in[1];
  const float* qkv_b  = (const float*)d_in[2];
  const float* proj_w = (const float*)d_in[3];
  const float* proj_b = (const float*)d_in[4];
  const float* reltab = (const float*)d_in[5];

  uint32_t* pk_qkv  = (uint32_t*)d_ws;
  uint32_t* pk_proj = pk_qkv + PK_QKV_DW;
  float*    addend  = (float*)(pk_proj + PK_PRJ_DW);

  int prep_total = PK_QKV_DW + PK_PRJ_DW + ADD_N;
  prep_tables<<<(prep_total + 255) / 256, 256, 0, stream>>>(
      qkv_w, proj_w, reltab, pk_qkv, pk_proj, addend);

  // one block per window: 64 images * 64 windows = 4096; 8 waves = 8 heads
  swin_window_attn<<<4096, 256, 128 * 1024, stream>>>(
      x, qkv_b, proj_b, pk_qkv, pk_proj, addend, (float*)d_out);
}
